// EditedMLPLayer_53558242181357
// MI455X (gfx1250) — matrix operand order repaired and checked by reading, not compiled, not run on hardware
//
#include <hip/hip_runtime.h>
#include <stdint.h>

// ---------------------------------------------------------------------------
// MI455X (gfx1250) EditedMLPLayer: bf16 WMMA GEMMs, fp32 accum, double-
// buffered LDS pipeline, async-to-LDS staging for the bf16 hidden operand.
// ---------------------------------------------------------------------------

typedef __attribute__((ext_vector_type(16))) __bf16 v16bf;
typedef __attribute__((ext_vector_type(8)))  __bf16 v8bf;
typedef __attribute__((ext_vector_type(8)))  float  v8f;

#define BM 128
#define BN 128
#define BK 32
#define LDS_STRIDE 40   // BK + 8 pad (bf16 elems); rows stay 16B aligned (80B)

union V16 { v16bf v; v8bf h[2]; };

__device__ __forceinline__ __bf16 f2bf(float f) {
  union { float f; uint32_t u; } c; c.f = f;
  uint32_t u = c.u + 0x7FFFu + ((c.u >> 16) & 1u);  // round-to-nearest-even
  uint16_t hs = (uint16_t)(u >> 16);
  __bf16 r; __builtin_memcpy(&r, &hs, 2); return r;
}

// Branchless tanh-GELU: tanh(u) = 1 - 2/(1+exp(2u)); exp via v_exp_f32.
__device__ __forceinline__ float gelu_fast(float x) {
  float u = 0.7978845608028654f * (x + 0.044715f * x * x * x);
  float e = __builtin_amdgcn_exp2f(2.8853900817779268f * u);   // exp(2u)
  float t = 1.0f - 2.0f * __builtin_amdgcn_rcpf(1.0f + e);     // saturates +-1
  return 0.5f * x * (1.0f + t);
}

// --- A tile (fp32 source): global -> regs, regs -> bf16 LDS ----------------
template <bool SCALED>
__device__ __forceinline__ void load_a_f32(float4* f, float& s, const float* A,
                                           long lda, int gm, int k0,
                                           const float* inv_rms, int tid) {
  int row  = tid >> 1;          // 0..127
  int half = (tid & 1) * 16;    // 0 or 16
  const float* src = A + (long)(gm + row) * lda + k0 + half;
  s = SCALED ? inv_rms[gm + row] : 1.0f;
  f[0] = ((const float4*)src)[0];
  f[1] = ((const float4*)src)[1];
  f[2] = ((const float4*)src)[2];
  f[3] = ((const float4*)src)[3];
}

__device__ __forceinline__ void store_a_f32(__bf16* sA, const float4* f, float s,
                                            int tid) {
  int row  = tid >> 1;
  int half = (tid & 1) * 16;
  v8bf lo, hi;
  lo[0]=f2bf(f[0].x*s); lo[1]=f2bf(f[0].y*s); lo[2]=f2bf(f[0].z*s); lo[3]=f2bf(f[0].w*s);
  lo[4]=f2bf(f[1].x*s); lo[5]=f2bf(f[1].y*s); lo[6]=f2bf(f[1].z*s); lo[7]=f2bf(f[1].w*s);
  hi[0]=f2bf(f[2].x*s); hi[1]=f2bf(f[2].y*s); hi[2]=f2bf(f[2].z*s); hi[3]=f2bf(f[2].w*s);
  hi[4]=f2bf(f[3].x*s); hi[5]=f2bf(f[3].y*s); hi[6]=f2bf(f[3].z*s); hi[7]=f2bf(f[3].w*s);
  *(v8bf*)&sA[row * LDS_STRIDE + half]     = lo;
  *(v8bf*)&sA[row * LDS_STRIDE + half + 8] = hi;
}

// --- A tile (bf16 hidden buffer): async DMA global -> LDS (ASYNCcnt) -------
// Per-lane: 32 contiguous bytes in both global and LDS; IOFFSET is added to
// both addresses, so two b128 ops at offset 0 / 16 cover the row segment.
__device__ __forceinline__ void async_a_bf16(const __bf16* sA, const uint16_t* A,
                                             long lda, int gm, int k0, int tid) {
  int row  = tid >> 1;
  int half = (tid & 1) * 16;
  const uint16_t* g = A + (long)(gm + row) * lda + k0 + half;
  // Generic LDS pointer: low 32 bits are the LDS byte address (ISA 10.2).
  uint32_t la = (uint32_t)(uintptr_t)(&sA[row * LDS_STRIDE + half]);
  asm volatile(
      "global_load_async_to_lds_b128 %0, %1, off\n\t"
      "global_load_async_to_lds_b128 %0, %1, off offset:16"
      :: "v"(la), "v"((unsigned long long)(uintptr_t)g)
      : "memory");
}

__device__ __forceinline__ void wait_async_lds() {
  asm volatile("s_wait_asynccnt 0" ::: "memory");
}

// --- B tile (fp32 source): global -> regs, regs -> transposed bf16 LDS -----
__device__ __forceinline__ void load_b_f32(float4* f, const float* Bsrc, long ldn,
                                           int k0, int gn, int tid) {
  int k    = tid >> 3;            // 0..31
  int nseg = (tid & 7) * 16;      // 0..112
  const float* src = Bsrc + (long)(k0 + k) * ldn + gn + nseg;
  f[0] = ((const float4*)src)[0];
  f[1] = ((const float4*)src)[1];
  f[2] = ((const float4*)src)[2];
  f[3] = ((const float4*)src)[3];
  __builtin_prefetch(src + (long)BK * ldn, 0, 0);   // next k-tile, same cols
}

__device__ __forceinline__ void store_b_f32(__bf16* sB, const float4* f, int tid) {
  int k    = tid >> 3;
  int nseg = (tid & 7) * 16;
#pragma unroll
  for (int j = 0; j < 4; ++j) {
    float4 q = f[j];
    int n0 = nseg + j * 4;
    sB[(n0 + 0) * LDS_STRIDE + k] = f2bf(q.x);
    sB[(n0 + 1) * LDS_STRIDE + k] = f2bf(q.y);
    sB[(n0 + 2) * LDS_STRIDE + k] = f2bf(q.z);
    sB[(n0 + 3) * LDS_STRIDE + k] = f2bf(q.w);
  }
}

// --- per-lane WMMA fragment loads per ISA layout tables --------------------
__device__ __forceinline__ v16bf load_a_frag(const __bf16* sA, int rowbase,
                                             int lhalf, int l15) {
  V16 r;
  int row = rowbase + l15;
  int kb  = lhalf * 8;
  r.h[0] = *(const v8bf*)&sA[row * LDS_STRIDE + kb];
  r.h[1] = *(const v8bf*)&sA[row * LDS_STRIDE + 16 + kb];
  return r.v;
}
__device__ __forceinline__ v16bf load_b_frag(const __bf16* sB, int colbase,
                                             int lhalf, int l15) {
  V16 r;
  int col = colbase + l15;
  int kb  = lhalf * 8;
  r.h[0] = *(const v8bf*)&sB[col * LDS_STRIDE + kb];
  r.h[1] = *(const v8bf*)&sB[col * LDS_STRIDE + 16 + kb];
  return r.v;
}

#define WMMA_BF16(a, b, c) \
  __builtin_amdgcn_wmma_f32_16x16x32_bf16(false, (a), false, (b), (short)0, (c), false, false)

__device__ __forceinline__ void wave_mma(const __bf16* sA, const __bf16* sB,
                                         int wm, int wn, int lhalf, int l15,
                                         v8f acc[4][2]) {
  v16bf af[4], bf[2];
#pragma unroll
  for (int mf = 0; mf < 4; ++mf) af[mf] = load_a_frag(sA, wm + mf * 16, lhalf, l15);
#pragma unroll
  for (int nf = 0; nf < 2; ++nf) bf[nf] = load_b_frag(sB, wn + nf * 16, lhalf, l15);
#pragma unroll
  for (int mf = 0; mf < 4; ++mf)
#pragma unroll
    for (int nf = 0; nf < 2; ++nf)
      acc[mf][nf] = WMMA_BF16(af[mf], bf[nf], acc[mf][nf]);
}

#define ZERO_ACC(acc)                                   \
  _Pragma("unroll") for (int mf = 0; mf < 4; ++mf)      \
  _Pragma("unroll") for (int nf = 0; nf < 2; ++nf)      \
  _Pragma("unroll") for (int i = 0; i < 8; ++i) acc[mf][nf][i] = 0.0f;

// ---------------------------------------------------------------------------
// Kernel 1: row inv-RMS of x
// ---------------------------------------------------------------------------
__global__ __launch_bounds__(256) void k_rms(const float* __restrict__ x,
                                             float* __restrict__ inv_rms, int D) {
  int row = blockIdx.x;
  const float* p = x + (long)row * D;
  float s = 0.0f;
  for (int c = threadIdx.x; c < D; c += 256) { float v = p[c]; s += v * v; }
#pragma unroll
  for (int m = 16; m >= 1; m >>= 1) s += __shfl_xor(s, m, 32);
  __shared__ float red[8];
  if ((threadIdx.x & 31) == 0) red[threadIdx.x >> 5] = s;
  __syncthreads();
  if (threadIdx.x == 0) {
    float t = 0.0f;
#pragma unroll
    for (int i = 0; i < 8; ++i) t += red[i];
    inv_rms[row] = rsqrtf(t / (float)D + 1e-6f);
  }
}

// ---------------------------------------------------------------------------
// Kernel 2: H = gelu(x @ W1 + b1), bf16 output. Double-buffered pipeline.
// ---------------------------------------------------------------------------
__global__ __launch_bounds__(256) void k_gemm1(const float* __restrict__ x,
                                               const float* __restrict__ W1,
                                               const float* __restrict__ b1,
                                               uint16_t* __restrict__ Hraw,
                                               int K, int N) {
  __shared__ __align__(16) __bf16 sA[2][BM * LDS_STRIDE];
  __shared__ __align__(16) __bf16 sB[2][BN * LDS_STRIDE];
  __bf16* H = (__bf16*)Hraw;
  int tid = threadIdx.x;
  int gm = blockIdx.y * BM, gn = blockIdx.x * BN;
  int w = tid >> 5, lane = tid & 31;
  int wm = (w & 1) * 64, wn = (w >> 1) * 32;
  int lhalf = lane >> 4, l15 = lane & 15;

  v8f acc[4][2];
  ZERO_ACC(acc)

  float4 fa[4], fb[4]; float s;
  load_a_f32<false>(fa, s, x, K, gm, 0, nullptr, tid);
  load_b_f32(fb, W1, N, 0, gn, tid);
  store_a_f32(sA[0], fa, s, tid);
  store_b_f32(sB[0], fb, tid);
  __syncthreads();

  int nIter = K / BK;
  for (int it = 0; it < nIter; ++it) {
    int cur = it & 1;
    bool more = (it + 1) < nIter;
    if (more) {
      load_a_f32<false>(fa, s, x, K, gm, (it + 1) * BK, nullptr, tid);
      load_b_f32(fb, W1, N, (it + 1) * BK, gn, tid);
    }
    wave_mma(sA[cur], sB[cur], wm, wn, lhalf, l15, acc);
    if (more) {
      store_a_f32(sA[cur ^ 1], fa, s, tid);
      store_b_f32(sB[cur ^ 1], fb, tid);
    }
    __syncthreads();
  }

#pragma unroll
  for (int nf = 0; nf < 2; ++nf) {
    int col = gn + wn + nf * 16 + l15;
    float bias = b1[col];
#pragma unroll
    for (int mf = 0; mf < 4; ++mf)
#pragma unroll
      for (int i = 0; i < 8; ++i) {
        int row = gm + wm + mf * 16 + lhalf * 8 + i;
        H[(long)row * N + col] = f2bf(gelu_fast(acc[mf][nf][i] + bias));
      }
  }
}

// ---------------------------------------------------------------------------
// Kernel 3: out = H(bf16) @ W2 + b2. A staged by async DMA to LDS.
// ---------------------------------------------------------------------------
__global__ __launch_bounds__(256) void k_gemm2(const uint16_t* __restrict__ H,
                                               const float* __restrict__ W2,
                                               const float* __restrict__ b2,
                                               float* __restrict__ out,
                                               int K, int N) {
  __shared__ __align__(16) __bf16 sA[2][BM * LDS_STRIDE];
  __shared__ __align__(16) __bf16 sB[2][BN * LDS_STRIDE];
  int tid = threadIdx.x;
  int gm = blockIdx.y * BM, gn = blockIdx.x * BN;
  int w = tid >> 5, lane = tid & 31;
  int wm = (w & 1) * 64, wn = (w >> 1) * 32;
  int lhalf = lane >> 4, l15 = lane & 15;

  v8f acc[4][2];
  ZERO_ACC(acc)

  float4 fb[4];
  async_a_bf16(sA[0], H, K, gm, 0, tid);
  load_b_f32(fb, W2, N, 0, gn, tid);
  store_b_f32(sB[0], fb, tid);
  wait_async_lds();
  __syncthreads();

  int nIter = K / BK;
  for (int it = 0; it < nIter; ++it) {
    int cur = it & 1;
    bool more = (it + 1) < nIter;
    if (more) {
      async_a_bf16(sA[cur ^ 1], H, K, gm, (it + 1) * BK, tid);
      load_b_f32(fb, W2, N, (it + 1) * BK, gn, tid);
    }
    wave_mma(sA[cur], sB[cur], wm, wn, lhalf, l15, acc);
    if (more) store_b_f32(sB[cur ^ 1], fb, tid);
    wait_async_lds();
    __syncthreads();
  }

#pragma unroll
  for (int nf = 0; nf < 2; ++nf) {
    int col = gn + wn + nf * 16 + l15;
    float bias = b2[col];
#pragma unroll
    for (int mf = 0; mf < 4; ++mf)
#pragma unroll
      for (int i = 0; i < 8; ++i) {
        int row = gm + wm + mf * 16 + lhalf * 8 + i;
        out[(long)row * N + col] = acc[mf][nf][i] + bias;
      }
  }
}

// ---------------------------------------------------------------------------
// Kernel 4: relu(rmsnorm(x) @ LoRA_A - kn_bias); per-row top-1 via packed
// (valbits<<32|idx) u64 atomicMax (relu => val>=0 => float bits monotone).
// ---------------------------------------------------------------------------
__global__ __launch_bounds__(256) void k_lora(const float* __restrict__ x,
                                              const float* __restrict__ LoRA_A,
                                              const float* __restrict__ kn_bias,
                                              const float* __restrict__ inv_rms,
                                              unsigned long long* __restrict__ packed,
                                              int K, int N) {
  __shared__ __align__(16) __bf16 sA[2][BM * LDS_STRIDE];
  __shared__ __align__(16) __bf16 sB[2][BN * LDS_STRIDE];
  int tid = threadIdx.x;
  int gm = blockIdx.y * BM, gn = blockIdx.x * BN;
  int w = tid >> 5, lane = tid & 31;
  int wm = (w & 1) * 64, wn = (w >> 1) * 32;
  int lhalf = lane >> 4, l15 = lane & 15;

  v8f acc[4][2];
  ZERO_ACC(acc)

  float4 fa[4], fb[4]; float s;
  load_a_f32<true>(fa, s, x, K, gm, 0, inv_rms, tid);
  load_b_f32(fb, LoRA_A, N, 0, gn, tid);
  store_a_f32(sA[0], fa, s, tid);
  store_b_f32(sB[0], fb, tid);
  __syncthreads();

  int nIter = K / BK;
  for (int it = 0; it < nIter; ++it) {
    int cur = it & 1;
    bool more = (it + 1) < nIter;
    if (more) {
      load_a_f32<true>(fa, s, x, K, gm, (it + 1) * BK, inv_rms, tid);
      load_b_f32(fb, LoRA_A, N, (it + 1) * BK, gn, tid);
    }
    wave_mma(sA[cur], sB[cur], wm, wn, lhalf, l15, acc);
    if (more) {
      store_a_f32(sA[cur ^ 1], fa, s, tid);
      store_b_f32(sB[cur ^ 1], fb, tid);
    }
    __syncthreads();
  }

  float kb0 = kn_bias[gn + wn + l15];
  float kb1 = kn_bias[gn + wn + 16 + l15];
#pragma unroll
  for (int mf = 0; mf < 4; ++mf)
#pragma unroll
    for (int i = 0; i < 8; ++i) {
      int rrow = gm + wm + mf * 16 + lhalf * 8 + i;
      float v0 = fmaxf(acc[mf][0][i] - kb0, 0.0f);
      float v1 = fmaxf(acc[mf][1][i] - kb1, 0.0f);
      float best; uint32_t bidx;
      if (v1 > v0) { best = v1; bidx = (uint32_t)(gn + wn + 16 + l15); }
      else         { best = v0; bidx = (uint32_t)(gn + wn + l15); }
      union { float f; uint32_t u; } cb; cb.f = best;
      uint32_t hi = cb.u, lo = bidx;
#pragma unroll
      for (int m = 1; m < 16; m <<= 1) {   // xor of bits 0..3 stays in half
        uint32_t ohi = __shfl_xor(hi, m, 32);
        uint32_t olo = __shfl_xor(lo, m, 32);
        if (ohi > hi || (ohi == hi && olo > lo)) { hi = ohi; lo = olo; }
      }
      if (l15 == 0) {
        unsigned long long p = ((unsigned long long)hi << 32) | lo;
        atomicMax(&packed[rrow], p);
      }
    }
}

// ---------------------------------------------------------------------------
// Kernel 5: out[row] += val * LoRA_B[idx]
// ---------------------------------------------------------------------------
__global__ __launch_bounds__(256) void k_edit(const unsigned long long* __restrict__ packed,
                                              const float* __restrict__ LoRA_B,
                                              float* __restrict__ out, int D) {
  int row = blockIdx.x;
  unsigned long long p = packed[row];
  union { uint32_t u; float f; } c; c.u = (uint32_t)(p >> 32);
  float val = c.f;
  int idx = (int)(uint32_t)(p & 0xFFFFFFFFu);
  const float* brow = LoRA_B + (long)idx * D;
  float* orow = out + (long)row * D;
  for (int cix = threadIdx.x; cix < D; cix += 256)
    orow[cix] += val * brow[cix];
}

// ---------------------------------------------------------------------------
extern "C" void kernel_launch(void* const* d_in, const int* in_sizes, int n_in,
                              void* d_out, int out_size, void* d_ws, size_t ws_size,
                              hipStream_t stream) {
  const float* x  = (const float*)d_in[0];
  const float* W1 = (const float*)d_in[1];
  const float* b1 = (const float*)d_in[2];
  const float* W2 = (const float*)d_in[3];
  const float* b2 = (const float*)d_in[4];
  const float* LA = (const float*)d_in[5];
  const float* LB = (const float*)d_in[6];
  const float* kb = (const float*)d_in[7];
  float* out = (float*)d_out;

  const int M = 4 * 2048;   // B*S
  const int D = 2048, F = 8192, R = 1024;

  char* ws = (char*)d_ws;
  uint16_t*           Hbf     = (uint16_t*)ws;                                  // M*F bf16
  float*              inv_rms = (float*)(ws + (size_t)M * F * 2);               // M f32
  unsigned long long* packed  = (unsigned long long*)(ws + (size_t)M * F * 2 + (size_t)M * 4);

  hipMemsetAsync(packed, 0, (size_t)M * sizeof(unsigned long long), stream);
  k_rms<<<M, 256, 0, stream>>>(x, inv_rms, D);
  k_gemm1<<<dim3(F / BN, M / BM), 256, 0, stream>>>(x, W1, b1, Hbf, D, F);
  k_gemm2<<<dim3(D / BN, M / BM), 256, 0, stream>>>(Hbf, W2, b2, out, F, D);
  k_lora<<<dim3(R / BN, M / BM), 256, 0, stream>>>(x, LA, kb, inv_rms, packed, D, R);
  k_edit<<<M, 256, 0, stream>>>(packed, LB, out, D);
}
